// RoPEMultiHeadAttention_50732153700826
// MI455X (gfx1250) — matrix operand order repaired and checked by reading, not compiled, not run on hardware
//
#include <hip/hip_runtime.h>
#include <hip/hip_bf16.h>

// ---------------------------------------------------------------------------
// CDNA5 (gfx1250) RoPE multi-head attention, bf16 WMMA pipeline with
// double-buffered async global->LDS staging.
// Shapes: B=1, S=2048, D=1024, H=16, hd=64.
// ---------------------------------------------------------------------------

typedef __attribute__((ext_vector_type(16))) __bf16 v16bf;
typedef __attribute__((ext_vector_type(8)))  float  v8f;
typedef __attribute__((ext_vector_type(4)))  int    v4i;

#define S_LEN   2048
#define D_MODEL 1024
#define NHEAD   16
#define HDIM    64

// ---- gfx1250 async global->LDS path (guarded; fallback = sync bounce) -----
#if defined(__has_builtin)
#if __has_builtin(__builtin_amdgcn_global_load_async_to_lds_b128)
#define HAVE_ASYNC_LOAD 1
#endif
#if __has_builtin(__builtin_amdgcn_s_wait_asynccnt)
#define HAVE_ASYNC_WAIT 1
#endif
#endif
#if defined(HAVE_ASYNC_LOAD) && defined(HAVE_ASYNC_WAIT)
#define USE_ASYNC_LDS 1
#endif

#ifdef USE_ASYNC_LDS
typedef __attribute__((address_space(1))) v4i as1_v4i;   // global int4
typedef __attribute__((address_space(3))) v4i as3_v4i;   // LDS int4
// Per-lane 16B copy: LDS[l] = MEM[g], tracked by ASYNCcnt (no VGPR data).
__device__ __forceinline__ void copy16_g2l(const __bf16* g, __bf16* l) {
    __builtin_amdgcn_global_load_async_to_lds_b128(
        (as1_v4i*)g, (as3_v4i*)l, 0, 0);
}
__device__ __forceinline__ void wait_async_lds() {
    __builtin_amdgcn_s_wait_asynccnt(0);
}
#else
__device__ __forceinline__ void copy16_g2l(const __bf16* g, __bf16* l) {
    *(uint4*)l = *(const uint4*)g;
}
__device__ __forceinline__ void wait_async_lds() {}
#endif

__device__ __forceinline__ v8f v8f_zero() {
    v8f z;
#pragma unroll
    for (int i = 0; i < 8; ++i) z[i] = 0.0f;
    return z;
}

__device__ __forceinline__ v16bf load_v16bf(const __bf16* p0, const __bf16* p1) {
    union { v16bf v; uint4 q[2]; } u;
    u.q[0] = *(const uint4*)p0;   // 8 contiguous bf16
    u.q[1] = *(const uint4*)p1;   // 8 contiguous bf16
    return u.v;
}

__device__ __forceinline__ v8f wmma_bf16(v16bf a, v16bf b, v8f c) {
    // D = A(16x32 bf16) * B(32x16 bf16) + C(16x16 f32)
    return __builtin_amdgcn_wmma_f32_16x16x32_bf16(
        false, a, false, b, (short)0, c, false, false);
}

// ---------------------------------------------------------------------------
// 1a) f32 -> bf16 conversion (same layout)
// ---------------------------------------------------------------------------
__global__ void __launch_bounds__(256)
cvt_f32_bf16(const float* __restrict__ in, __bf16* __restrict__ out, int n) {
    int i = blockIdx.x * 256 + threadIdx.x;
    if (i < n) out[i] = (__bf16)in[i];
}

// ---------------------------------------------------------------------------
// 1b) f32 -> bf16 with transpose: in[N][K] -> out[K][N]
//     (one-time pass so GEMM B-staging becomes a straight async copy)
// ---------------------------------------------------------------------------
__global__ void __launch_bounds__(256)
cvt_f32_bf16_t(const float* __restrict__ in, __bf16* __restrict__ out,
               int N, int K) {
    int i = blockIdx.x * 256 + threadIdx.x;
    if (i < N * K) {
        int n = i / K;
        int k = i - n * K;
        out[(size_t)k * N + n] = (__bf16)in[i];
    }
}

// ---------------------------------------------------------------------------
// 2) Tiled bf16 WMMA GEMM:  C[M][N] = A[M][Kd] * Bt[Kd][N] + bias[N]
//    Block: 256 thr (8 waves), 128x128 tile, Kd staged in 64-chunks,
//    double-buffered async global->LDS staging overlapped with WMMA.
// ---------------------------------------------------------------------------
#define BM 128
#define BN 128
#define BK 64

__global__ void __launch_bounds__(256)
gemm_bf16_wmma(const __bf16* __restrict__ A,   // [M][Kd]
               const __bf16* __restrict__ Bt,  // [Kd][N]
               const float*  __restrict__ bias,
               float*        __restrict__ C,   // [M][N]
               int M, int N, int Kd)
{
    __shared__ __align__(16) __bf16 As [2][BM * BK];  // [m][k]
    __shared__ __align__(16) __bf16 Bts[2][BK * BN];  // [k][n]

    const int tid  = threadIdx.x;
    const int lane = tid & 31;
    const int wave = tid >> 5;
    const int half = lane >> 4;
    const int ml   = lane & 15;
    const int m_blk = blockIdx.y * BM;
    const int n_blk = blockIdx.x * BN;
    const int wm = (wave >> 1) * 32;   // 0,32,64,96
    const int wn = (wave & 1) * 64;    // 0,64

    v8f acc[2][4];
#pragma unroll
    for (int i = 0; i < 2; ++i)
#pragma unroll
        for (int j = 0; j < 4; ++j) acc[i][j] = v8f_zero();

    // Stage one BKxBM A-tile + BKxBN Bt-tile into buffer `b` (async issue only)
    auto stage = [&](int b, int k0) {
#pragma unroll
        for (int it = 0; it < 4; ++it) {
            int idx = (it * 256 + tid) * 8;
            int r = idx >> 6;               // A row   (/BK)
            int c = idx & (BK - 1);         // A col
            copy16_g2l(&A[(size_t)(m_blk + r) * Kd + k0 + c], &As[b][idx]);
        }
#pragma unroll
        for (int it = 0; it < 4; ++it) {
            int idx = (it * 256 + tid) * 8;
            int k = idx >> 7;               // Bt row  (/BN)
            int n = idx & (BN - 1);         // Bt col
            copy16_g2l(&Bt[(size_t)(k0 + k) * N + n_blk + n], &Bts[b][idx]);
        }
    };

    int buf = 0;
    stage(buf, 0);
    wait_async_lds();
    __syncthreads();

    for (int k0 = 0; k0 < Kd; k0 += BK) {
        // Kick off next tile's DMA while we compute on the current one.
        if (k0 + BK < Kd) stage(buf ^ 1, k0 + BK);

#pragma unroll
        for (int kc = 0; kc < BK; kc += 32) {
            v16bf a[2], b[4];
            // A-layout (16x32): lane row m=ml, k = (j>>3)*16 + half*8 + (j&7)
#pragma unroll
            for (int i = 0; i < 2; ++i) {
                const __bf16* base =
                    &As[buf][(wm + i * 16 + ml) * BK + kc + half * 8];
                a[i] = load_v16bf(base, base + 16);
            }
            // B-layout (32x16): lane column n = ml, k = (s>>3)*16 + half*8 + (s&7)
#pragma unroll
            for (int j = 0; j < 4; ++j) {
#pragma unroll
                for (int s = 0; s < 16; ++s)
                    b[j][s] = Bts[buf][(kc + (s >> 3) * 16 + half * 8 + (s & 7)) * BN + wn + j * 16 + ml];
            }
#pragma unroll
            for (int i = 0; i < 2; ++i)
#pragma unroll
                for (int j = 0; j < 4; ++j)
                    acc[i][j] = wmma_bf16(a[i], b[j], acc[i][j]);
        }

        wait_async_lds();    // next buffer's DMA complete (own wave)
        __syncthreads();     // all waves done computing + DMA landed
        buf ^= 1;
    }

    // D-layout store: lane holds N=ml, rows M = half*8 + r
#pragma unroll
    for (int i = 0; i < 2; ++i) {
        int mbase = m_blk + wm + i * 16 + half * 8;
#pragma unroll
        for (int j = 0; j < 4; ++j) {
            int col = n_blk + wn + j * 16 + ml;
            float bv = bias[col];
#pragma unroll
            for (int r = 0; r < 8; ++r)
                C[(size_t)(mbase + r) * N + col] = acc[i][j][r] + bv;
        }
    }
}

// ---------------------------------------------------------------------------
// 3) RoPE + head split: qkv f32 [S][3*1024] ->
//    qb [H][S][64], kt [H][64][S] (feature-major!), vb [H][S][64]; bf16.
// ---------------------------------------------------------------------------
__global__ void __launch_bounds__(256)
rope_split(const float* __restrict__ qkv,
           __bf16* __restrict__ qb, __bf16* __restrict__ kt,
           __bf16* __restrict__ vb)
{
    int i = blockIdx.x * 256 + threadIdx.x;       // H*S*64 threads exactly
    int d = i & (HDIM - 1);
    int s = (i >> 6) & (S_LEN - 1);
    int h = i >> 17;                               // /(S*64)

    const float* row = qkv + (size_t)s * (3 * D_MODEL);
    float q = row[h * HDIM + d];
    float k = row[D_MODEL + h * HDIM + d];
    float v = row[2 * D_MODEL + h * HDIM + d];

    int di = d & 31;
    float inv = __powf(10000.0f, -(float)di / 32.0f);
    float sn, c;
    __sincosf((float)s * inv, &sn, &c);   // sin first, cos second
    float qr, kr;
    if (d < 32) {
        qr = -row[h * HDIM + d + 32];
        kr = -row[D_MODEL + h * HDIM + d + 32];
    } else {
        qr =  row[h * HDIM + d - 32];
        kr =  row[D_MODEL + h * HDIM + d - 32];
    }
    size_t o = ((size_t)h * S_LEN + s) * HDIM + d;
    qb[o] = (__bf16)(q * c + qr * sn);
    vb[o] = (__bf16)v;
    // K feature-major so flash staging is a straight (async) copy
    kt[((size_t)h * HDIM + d) * S_LEN + s] = (__bf16)(k * c + kr * sn);
}

// ---------------------------------------------------------------------------
// 4) Flash attention (bf16 WMMA, f32 online softmax).
//    Grid (H, S/128). Block = 256 thr (8 waves); wave owns 16 query rows.
//    64-key blocks double-buffered in LDS via async DMA:
//      Kt [feat][key] (from feature-major k), V [key][feat].
// ---------------------------------------------------------------------------
__global__ void __launch_bounds__(256)
flash_attn(const __bf16* __restrict__ qb,
           const __bf16* __restrict__ ktg,   // [H][64][S]
           const __bf16* __restrict__ vb,
           __bf16* __restrict__ ctx)         // [S][D] bf16
{
    __shared__ __align__(16) __bf16 Kt[2][HDIM * 64];   // [feat][key]
    __shared__ __align__(16) __bf16 Vs[2][64 * HDIM];   // [key][feat]
    __shared__ __align__(16) __bf16 Ps[8][16 * 64];     // per-wave P tile

    const int tid  = threadIdx.x;
    const int lane = tid & 31;
    const int wave = tid >> 5;
    const int half = lane >> 4;
    const int ml   = lane & 15;
    const int h  = blockIdx.x;
    const int s0 = blockIdx.y * 128 + wave * 16;

    const __bf16* qh = qb  + (size_t)h * S_LEN * HDIM;
    const __bf16* kh = ktg + (size_t)h * HDIM * S_LEN;
    const __bf16* vh = vb  + (size_t)h * S_LEN * HDIM;

    // Q in WMMA A-layout, 2 k-steps over hd=64
    v16bf qa[2];
#pragma unroll
    for (int ks = 0; ks < 2; ++ks) {
        const __bf16* base = qh + (size_t)(s0 + ml) * HDIM + ks * 32 + half * 8;
        qa[ks] = load_v16bf(base, base + 16);
    }

    v8f acc[4];
#pragma unroll
    for (int j = 0; j < 4; ++j) acc[j] = v8f_zero();
    float mrow[8], lrow[8];
#pragma unroll
    for (int r = 0; r < 8; ++r) { mrow[r] = -1e30f; lrow[r] = 0.0f; }
    const float scale = 0.125f;   // 1/sqrt(64)

    // Async stage of one 64-key block into buffer `b` (issue only)
    auto stage = [&](int b, int kb0) {
#pragma unroll
        for (int it = 0; it < 2; ++it) {
            int idx = (it * 256 + tid) * 8;
            int r = idx >> 6;               // Kt: feat | Vs: key
            int c = idx & 63;               // Kt: key  | Vs: feat
            copy16_g2l(kh + (size_t)r * S_LEN + kb0 + c, &Kt[b][idx]);
            copy16_g2l(vh + (size_t)(kb0 + r) * HDIM + c, &Vs[b][idx]);
        }
    };

    int buf = 0;
    stage(buf, 0);
    wait_async_lds();
    __syncthreads();

    for (int kb0 = 0; kb0 < S_LEN; kb0 += 64) {
        // DMA next key block while computing on this one
        if (kb0 + 64 < S_LEN) stage(buf ^ 1, kb0 + 64);

        // Scores: 16 x 64 (4 sub-tiles), S = Q * K^T
        v8f sc[4];
#pragma unroll
        for (int j = 0; j < 4; ++j) {
            sc[j] = v8f_zero();
#pragma unroll
            for (int ks = 0; ks < 2; ++ks) {
                v16bf kb;
#pragma unroll
                for (int s = 0; s < 16; ++s)
                    kb[s] = Kt[buf][(ks * 32 + (s >> 3) * 16 + half * 8 + (s & 7)) * 64 + j * 16 + ml];
                sc[j] = wmma_bf16(qa[ks], kb, sc[j]);
            }
        }

        // Online softmax (row-wise over 64 keys). D-layout: row M=half*8+r
        float newmax[8];
#pragma unroll
        for (int r = 0; r < 8; ++r) {
            float v = mrow[r];
#pragma unroll
            for (int j = 0; j < 4; ++j) v = fmaxf(v, sc[j][r] * scale);
#pragma unroll
            for (int off = 1; off < 16; off <<= 1)
                v = fmaxf(v, __shfl_xor(v, off));
            newmax[r] = v;
        }
#pragma unroll
        for (int r = 0; r < 8; ++r) {
            float corr = __expf(mrow[r] - newmax[r]);
            mrow[r] = newmax[r];
            lrow[r] *= corr;
#pragma unroll
            for (int j = 0; j < 4; ++j) acc[j][r] *= corr;
        }
        float psum[8];
#pragma unroll
        for (int r = 0; r < 8; ++r) psum[r] = 0.0f;
#pragma unroll
        for (int j = 0; j < 4; ++j) {
#pragma unroll
            for (int r = 0; r < 8; ++r) {
                float p = __expf(sc[j][r] * scale - mrow[r]);
                psum[r] += p;
                Ps[wave][(half * 8 + r) * 64 + j * 16 + ml] = (__bf16)p;
            }
        }
#pragma unroll
        for (int r = 0; r < 8; ++r) {
            float v = psum[r];
#pragma unroll
            for (int off = 1; off < 16; off <<= 1) v += __shfl_xor(v, off);
            lrow[r] += v;
        }
        __syncthreads();   // Ps store->load ordering (uniform across waves)

        // ctx += P(16x64) * V(64x64): 2 k-slabs of 32 keys, 4 feat tiles
#pragma unroll
        for (int ksl = 0; ksl < 2; ++ksl) {
            const __bf16* abase = &Ps[wave][ml * 64 + ksl * 32 + half * 8];
            v16bf pa = load_v16bf(abase, abase + 16);
#pragma unroll
            for (int j = 0; j < 4; ++j) {
                v16bf vbv;
#pragma unroll
                for (int s = 0; s < 16; ++s)
                    vbv[s] = Vs[buf][(ksl * 32 + (s >> 3) * 16 + half * 8 + (s & 7)) * HDIM + j * 16 + ml];
                acc[j] = wmma_bf16(pa, vbv, acc[j]);
            }
        }

        wait_async_lds();    // next buffer's DMA complete (own wave)
        __syncthreads();     // all waves finished reads + DMA landed
        buf ^= 1;
    }

    // Epilogue: normalize, write ctx bf16 at [s][h*64 + f]
#pragma unroll
    for (int j = 0; j < 4; ++j) {
#pragma unroll
        for (int r = 0; r < 8; ++r) {
            float o = acc[j][r] / lrow[r];
            int srow = s0 + half * 8 + r;
            int col  = h * HDIM + j * 16 + ml;
            ctx[(size_t)srow * D_MODEL + col] = (__bf16)o;
        }
    }
}

// ---------------------------------------------------------------------------
// Launch
// ---------------------------------------------------------------------------
extern "C" void kernel_launch(void* const* d_in, const int* in_sizes, int n_in,
                              void* d_out, int out_size, void* d_ws, size_t ws_size,
                              hipStream_t stream)
{
    const float* query = (const float*)d_in[0];
    // d_in[1] (key), d_in[2] (value) are unused by the reference module.
    const float* qkv_w = (const float*)d_in[3];
    const float* qkv_b = (const float*)d_in[4];
    const float* out_w = (const float*)d_in[5];
    const float* out_b = (const float*)d_in[6];
    float* out = (float*)d_out;

    char* ws = (char*)d_ws;
    size_t off = 0;
    auto alloc = [&](size_t bytes) {
        size_t o = off;
        off += (bytes + 255) & ~(size_t)255;
        return o;
    };
    __bf16* q_b16   = (__bf16*)(ws + alloc((size_t)S_LEN * D_MODEL * 2));
    __bf16* wt_b16  = (__bf16*)(ws + alloc((size_t)3 * D_MODEL * D_MODEL * 2)); // [K][3D]
    __bf16* owt_b16 = (__bf16*)(ws + alloc((size_t)D_MODEL * D_MODEL * 2));     // [K][D]
    float*  qkv_f32 = (float*) (ws + alloc((size_t)S_LEN * 3 * D_MODEL * 4));
    __bf16* q_rope  = (__bf16*)(ws + alloc((size_t)S_LEN * D_MODEL * 2));
    __bf16* kt_rope = (__bf16*)(ws + alloc((size_t)S_LEN * D_MODEL * 2));       // [H][64][S]
    __bf16* v_bf    = (__bf16*)(ws + alloc((size_t)S_LEN * D_MODEL * 2));
    __bf16* ctx_b16 = (__bf16*)(ws + alloc((size_t)S_LEN * D_MODEL * 2));

    // 1) precision convert; weights transposed to [K][N] for async staging
    {
        int n1 = S_LEN * D_MODEL;
        int n2 = 3 * D_MODEL * D_MODEL;
        int n3 = D_MODEL * D_MODEL;
        cvt_f32_bf16<<<(n1 + 255) / 256, 256, 0, stream>>>(query, q_b16, n1);
        cvt_f32_bf16_t<<<(n2 + 255) / 256, 256, 0, stream>>>(
            qkv_w, wt_b16, 3 * D_MODEL, D_MODEL);
        cvt_f32_bf16_t<<<(n3 + 255) / 256, 256, 0, stream>>>(
            out_w, owt_b16, D_MODEL, D_MODEL);
    }

    // 2) QKV projection: [2048x1024] @ Bt[1024][3072] + bias -> f32
    gemm_bf16_wmma<<<dim3((3 * D_MODEL) / BN, S_LEN / BM), 256, 0, stream>>>(
        q_b16, wt_b16, qkv_b, qkv_f32, S_LEN, 3 * D_MODEL, D_MODEL);

    // 3) RoPE + per-head split to bf16 (K emitted feature-major)
    rope_split<<<(NHEAD * S_LEN * HDIM) / 256, 256, 0, stream>>>(
        qkv_f32, q_rope, kt_rope, v_bf);

    // 4) Flash attention
    flash_attn<<<dim3(NHEAD, S_LEN / 128), 256, 0, stream>>>(
        q_rope, kt_rope, v_bf, ctx_b16);

    // 5) Output projection: [2048x1024] @ Bt[1024][1024] + bias -> f32 out
    gemm_bf16_wmma<<<dim3(D_MODEL / BN, S_LEN / BM), 256, 0, stream>>>(
        ctx_b16, owt_b16, out_b, out, S_LEN, D_MODEL, D_MODEL);
}
